// HRM_L_Module_3573412790569
// MI455X (gfx1250) — matrix operand order repaired and checked by reading, not compiled, not run on hardware
//
#include <hip/hip_runtime.h>
#include <hip/hip_bf16.h>
#include <math.h>

// ---------------------------------------------------------------------------
// Sizes from reference
// ---------------------------------------------------------------------------
#define BB 64
#define SS 2048
#define EE 1024
#define DD 1024
#define SYNC 512
#define MM 32
#define HEADS 16
#define DH 64
#define SYN_H 2048
#define TR_H 64

typedef __bf16 bf16_t;
typedef __attribute__((ext_vector_type(16))) __bf16 v16bf;
typedef __attribute__((ext_vector_type(8)))  __bf16 v8bf;
typedef __attribute__((ext_vector_type(8)))  float  v8f;

union V16B { v16bf v; v8bf h[2]; };

// ---------------------------------------------------------------------------
// Workspace layout (bytes)
// ---------------------------------------------------------------------------
static constexpr size_t OFF_QVEC = 0;                                   // B*E f32
static constexpr size_t OFF_PQ   = OFF_QVEC + (size_t)BB*EE*4;          // B*E f32
static constexpr size_t OFF_WKB  = OFF_PQ   + (size_t)BB*EE*4;          // E*E bf16
static constexpr size_t OFF_WVB  = OFF_WKB  + (size_t)EE*EE*2;          // E*E bf16
static constexpr size_t OFF_KVS  = OFF_WVB  + (size_t)EE*EE*2;          // B*H*64*64 f32
static constexpr size_t OFF_KSUM = OFF_KVS  + (size_t)BB*HEADS*DH*DH*4; // B*H*64 f32
static constexpr size_t OFF_ATTN = OFF_KSUM + (size_t)BB*HEADS*DH*4;    // B*E f32
static constexpr size_t OFF_AO   = OFF_ATTN + (size_t)BB*EE*4;          // B*E f32
static constexpr size_t OFF_H1   = OFF_AO   + (size_t)BB*EE*4;          // B*SYN_H f32
static constexpr size_t OFF_ST   = OFF_H1   + (size_t)BB*SYN_H*4;       // B*D f32

// ---------------------------------------------------------------------------
// K1a: q = sync_action @ qp_w + qp_b          (B x E, tiny)
// ---------------------------------------------------------------------------
__global__ __launch_bounds__(256) void qproj_kernel(
    const float* __restrict__ sync_action, const float* __restrict__ qp_w,
    const float* __restrict__ qp_b, float* __restrict__ qvec) {
  int idx = blockIdx.x * 256 + threadIdx.x;          // B*E threads
  int b = idx >> 10, n = idx & 1023;
  const float* s = sync_action + (size_t)b * SYNC;
  float acc = qp_b[n];
  for (int k = 0; k < SYNC; ++k) acc = fmaf(s[k], qp_w[(size_t)k * EE + n], acc);
  qvec[idx] = acc;
}

// ---------------------------------------------------------------------------
// K1b: pq = phi(q @ wq + bq)                  (B x E, tiny)
// ---------------------------------------------------------------------------
__global__ __launch_bounds__(256) void pq_kernel(
    const float* __restrict__ qvec, const float* __restrict__ wq,
    const float* __restrict__ bq, float* __restrict__ pq) {
  int idx = blockIdx.x * 256 + threadIdx.x;
  int b = idx >> 10, n = idx & 1023;
  const float* qv = qvec + (size_t)b * EE;
  float acc = bq[n];
  for (int k = 0; k < EE; ++k) acc = fmaf(qv[k], wq[(size_t)k * EE + n], acc);
  pq[idx] = (acc > 0.f) ? (acc + 1.f) : __expf(acc);   // elu(x)+1
}

// ---------------------------------------------------------------------------
// K2: convert wk/wv f32 -> bf16 in WMMA B-fragment tile layout:
//   tile (nt 0..63, kt 0..31): 32 lanes x 16 contiguous bf16 per lane.
//   lane: n = nt*16 + (lane&15);  k = kt*32 + (lane>>4)*16 + i
// ---------------------------------------------------------------------------
__global__ __launch_bounds__(128) void conv_weight_kernel(
    const float* __restrict__ wk, const float* __restrict__ wv,
    bf16_t* __restrict__ wkb, bf16_t* __restrict__ wvb) {
  int tile = blockIdx.x * 4 + (threadIdx.x >> 5);     // 0..4095 (2 weights x 2048 tiles)
  int lane = threadIdx.x & 31;
  const float* src = (tile < 2048) ? wk : wv;
  bf16_t*      dst = (tile < 2048) ? wkb : wvb;
  int local = tile & 2047;
  int nt = local >> 5, kt = local & 31;
  int n  = nt * 16 + (lane & 15);
  int k0 = kt * 32 + ((lane >> 4) * 16);
  bf16_t* d = dst + ((size_t)local * 32 + lane) * 16;
  for (int i = 0; i < 16; ++i) d[i] = (bf16_t)src[(size_t)(k0 + i) * EE + n];
}

// ---------------------------------------------------------------------------
// K3: dominant fused kernel (bf16 WMMA, f32 accumulate).
// One workgroup per (b, head-pair); 8 waves. Wave -> (head, K/V, n-pair);
// each wave owns a 2x2 (s-tile x n-tile) block so every A *and* B fragment
// feeds 2 WMMAs (halves L2 weight traffic vs 1x4 mapping).
// Per 32-row S-chunk:
//   kvb  = bf16(x_context + zH)                    -> LDS (two 512-col halves)
//   kh/vh (32x64) = kvb @ w{k,v}[:, h*64:h*64+64]  via v_wmma_f32_16x16x32_bf16
//   pk = phi(kh + bk), vh += bv; transpose-store bf16 to LDS
//   kv_state(64x64) += pk^T @ vh                   via 4 more WMMAs / wave
//   ksum += colsum(pk)
// Prefetches the next S-chunk while converting the current one.
// ---------------------------------------------------------------------------
__global__ __launch_bounds__(256) void kv_state_kernel(
    const float* __restrict__ x_context, const float* __restrict__ zH,
    const float* __restrict__ bk, const float* __restrict__ bv,
    const bf16_t* __restrict__ wkb, const bf16_t* __restrict__ wvb,
    float* __restrict__ kv_state, float* __restrict__ ksum) {
  __shared__ bf16_t kvb[32 * 512];      // 32 rows x 512-col half of kv tile (32 KB)
  __shared__ bf16_t pkT[2][64 * 32];    // per-head phi(kh)^T : [d][s]  (2 x 4 KB)
  __shared__ bf16_t vhT[2][64 * 32];    // per-head vh^T      : [d][s]  (2 x 4 KB)

  const int b    = blockIdx.x >> 3;     // 8 head-pair blocks per batch
  const int hp   = blockIdx.x & 7;      // head pair index
  const int tid  = threadIdx.x;
  const int wave = tid >> 5;            // 0..7
  const int hl   = wave >> 2;           // local head (0/1)
  const int h    = hp * 2 + hl;         // global head
  const int isV  = (wave >> 1) & 1;     // 0: K-projection, 1: V-projection
  const int np   = wave & 1;            // n-pair: n-tiles {2np, 2np+1}
  const int w4   = wave & 3;            // wave within head (phase-3 stripe)
  const int lane = tid & 31;
  const int l15  = lane & 15;
  const int lhi  = lane >> 4;

  v8f acc_kv[4];
  for (int j = 0; j < 4; ++j) acc_kv[j] = {};
  float ks_acc = 0.f;                   // tid<128: head tid>>6, d = tid&63

  const float* xb   = x_context + (size_t)b * SS * EE;
  const float* zhb  = zH + (size_t)b * EE;
  const bf16_t* wsrc = isV ? wvb : wkb;

  for (int s0 = 0; s0 < SS; s0 += 32) {
    v8f acc2[4];                        // [st*2 + ntl]
    for (int j = 0; j < 4; ++j) acc2[j] = {};

    for (int half = 0; half < 2; ++half) {
      // ---- phase 1: load 32x512 f32, add zH, cvt bf16 -> LDS (coalesced) --
      for (int i = 0; i < 16; ++i) {
        int flat = i * 256 + tid;
        int flat4 = flat * 4;
        int row = flat4 >> 9;
        int ch  = flat4 & 511;
        int col = half * 512 + ch;
        const float* xp = xb + (size_t)(s0 + row) * EE + col;
        const float4 xv = *(const float4*)xp;
        const float4 zv = *(const float4*)(zhb + col);
        bf16_t* dst = &kvb[row * 512 + ch];
        dst[0] = (bf16_t)(xv.x + zv.x); dst[1] = (bf16_t)(xv.y + zv.y);
        dst[2] = (bf16_t)(xv.z + zv.z); dst[3] = (bf16_t)(xv.w + zv.w);
        // prefetch next S-chunk (one request per 128B line)
        if ((s0 + 32 < SS) && ((flat & 7) == 0))
          __builtin_prefetch(xp + 32 * EE, 0, 1);
      }
      __syncthreads();

      // ---- phase 2: 2x2 register blocking; each A/B fragment -> 2 WMMAs --
#pragma unroll 2
      for (int et = 0; et < 16; ++et) {
        V16B a0, a1;
        const bf16_t* ap0 = &kvb[l15 * 512 + et * 32 + lhi * 16];         // st=0
        const bf16_t* ap1 = &kvb[(16 + l15) * 512 + et * 32 + lhi * 16];  // st=1
        a0.h[0] = *(const v8bf*)ap0;  a0.h[1] = *(const v8bf*)(ap0 + 8);
        a1.h[0] = *(const v8bf*)ap1;  a1.h[1] = *(const v8bf*)(ap1 + 8);
        const int ktg = half * 16 + et;
        const v16bf b0 = *(const v16bf*)(
            wsrc + ((size_t)((h * 4 + np * 2 + 0) * 32 + ktg) * 32 + lane) * 16);
        const v16bf b1 = *(const v16bf*)(
            wsrc + ((size_t)((h * 4 + np * 2 + 1) * 32 + ktg) * 32 + lane) * 16);
        acc2[0] = __builtin_amdgcn_wmma_f32_16x16x32_bf16(
            false, a0.v, false, b0, (short)0, acc2[0], false, false);
        acc2[1] = __builtin_amdgcn_wmma_f32_16x16x32_bf16(
            false, a0.v, false, b1, (short)0, acc2[1], false, false);
        acc2[2] = __builtin_amdgcn_wmma_f32_16x16x32_bf16(
            false, a1.v, false, b0, (short)0, acc2[2], false, false);
        acc2[3] = __builtin_amdgcn_wmma_f32_16x16x32_bf16(
            false, a1.v, false, b1, (short)0, acc2[3], false, false);
      }
      __syncthreads();
    }

    // ---- epilogue: bias (+phi for kh), transpose-store bf16 to LDS --------
    {
      bf16_t* dstT = isV ? vhT[hl] : pkT[hl];
      int mbase = (lane < 16) ? 0 : 8;
      for (int stl = 0; stl < 2; ++stl)
        for (int ntl = 0; ntl < 2; ++ntl) {
          int nt = np * 2 + ntl;
          int ncol = h * 64 + nt * 16 + l15;
          float bias = isV ? bv[ncol] : bk[ncol];
          int dloc = nt * 16 + l15;              // d within head: 0..63
          for (int r = 0; r < 8; ++r) {
            float val = acc2[stl * 2 + ntl][r] + bias;
            if (!isV) val = (val > 0.f) ? (val + 1.f) : __expf(val);   // phi
            int srow = stl * 16 + mbase + r;
            dstT[dloc * 32 + srow] = (bf16_t)val;
          }
        }
    }
    __syncthreads();

    // ---- ksum accumulation (pk column sums over this chunk) ---------------
    if (tid < 128) {
      int hk = tid >> 6, d = tid & 63;
      float s = 0.f;
      for (int k = 0; k < 32; ++k) s += (float)pkT[hk][d * 32 + k];
      ks_acc += s;
    }

    // ---- phase 3: kv_state(64x64) += pk^T @ vh  (K=32, 4 WMMAs/wave) ------
    {
      V16B a;
      const bf16_t* ap = &pkT[hl][(w4 * 16 + l15) * 32 + lhi * 16];
      a.h[0] = *(const v8bf*)ap;
      a.h[1] = *(const v8bf*)(ap + 8);
      for (int j = 0; j < 4; ++j) {
        const bf16_t* bp = &vhT[hl][(j * 16 + l15) * 32 + lhi * 16];
        v16bf bmat = *(const v16bf*)bp;
        acc_kv[j] = __builtin_amdgcn_wmma_f32_16x16x32_bf16(
            false, a.v, false, bmat, (short)0, acc_kv[j], false, false);
      }
    }
    __syncthreads();
  }

  // ---- writeback: kv_state [b][h][dk][dv] f32, ksum [b][h][d] -------------
  float* kvs = kv_state + (size_t)(b * HEADS + h) * DH * DH;
  int mbase = (lane < 16) ? 0 : 8;
  for (int j = 0; j < 4; ++j)
    for (int r = 0; r < 8; ++r) {
      int drow = w4 * 16 + mbase + r;
      int dcol = j * 16 + l15;
      kvs[drow * DH + dcol] = acc_kv[j][r];
    }
  if (tid < 128)
    ksum[(size_t)(b * HEADS + hp * 2 + (tid >> 6)) * DH + (tid & 63)] = ks_acc;
}

// ---------------------------------------------------------------------------
// K4a: attn[b, h*64+e] = (pq . kv_state[:,e]) / (pq . ksum + 1e-6)
// ---------------------------------------------------------------------------
__global__ __launch_bounds__(256) void attn_kernel(
    const float* __restrict__ pq, const float* __restrict__ kv_state,
    const float* __restrict__ ksum, float* __restrict__ attn) {
  int idx = blockIdx.x * 256 + threadIdx.x;           // B*H*64
  int b = idx >> 10, h = (idx >> 6) & 15, e = idx & 63;
  const float* pqv = pq + (size_t)b * EE + h * DH;
  const float* kvs = kv_state + (size_t)(b * HEADS + h) * DH * DH;
  const float* ksv = ksum + (size_t)(b * HEADS + h) * DH;
  float num = 0.f, den = 0.f;
  for (int d = 0; d < DH; ++d) {
    float p = pqv[d];
    num = fmaf(p, kvs[d * DH + e], num);
    den = fmaf(p, ksv[d], den);
  }
  attn[idx] = num / (den + 1e-6f);
}

// ---------------------------------------------------------------------------
// K4b: attn_out = attn @ wo + bo
// ---------------------------------------------------------------------------
__global__ __launch_bounds__(256) void attnout_kernel(
    const float* __restrict__ attn, const float* __restrict__ wo,
    const float* __restrict__ bo, float* __restrict__ attn_out) {
  int idx = blockIdx.x * 256 + threadIdx.x;           // B*E
  int b = idx >> 10, n = idx & 1023;
  const float* av = attn + (size_t)b * EE;
  float acc = bo[n];
  for (int k = 0; k < EE; ++k) acc = fmaf(av[k], wo[(size_t)k * EE + n], acc);
  attn_out[idx] = acc;
}

// ---------------------------------------------------------------------------
// K5a: h1 = relu(concat(attn_out, activated_zL) @ syn_w1 + syn_b1)
// ---------------------------------------------------------------------------
__global__ __launch_bounds__(256) void syn1_kernel(
    const float* __restrict__ attn_out, const float* __restrict__ azL,
    const float* __restrict__ syn_w1, const float* __restrict__ syn_b1,
    float* __restrict__ h1) {
  int idx = blockIdx.x * 256 + threadIdx.x;           // B*SYN_H
  int b = idx >> 11, j = idx & 2047;
  const float* a0 = attn_out + (size_t)b * EE;
  const float* a1 = azL + (size_t)b * DD;
  float acc = syn_b1[j];
  for (int k = 0; k < EE; ++k) acc = fmaf(a0[k], syn_w1[(size_t)k * SYN_H + j], acc);
  for (int k = 0; k < DD; ++k) acc = fmaf(a1[k], syn_w1[(size_t)(EE + k) * SYN_H + j], acc);
  h1[idx] = fmaxf(acc, 0.f);
}

// ---------------------------------------------------------------------------
// K5b: state = h1 @ syn_w2 + syn_b2
// ---------------------------------------------------------------------------
__global__ __launch_bounds__(256) void syn2_kernel(
    const float* __restrict__ h1, const float* __restrict__ syn_w2,
    const float* __restrict__ syn_b2, float* __restrict__ state) {
  int idx = blockIdx.x * 256 + threadIdx.x;           // B*D
  int b = idx >> 10, n = idx & 1023;
  const float* hv = h1 + (size_t)b * SYN_H;
  float acc = syn_b2[n];
  for (int k = 0; k < SYN_H; ++k) acc = fmaf(hv[k], syn_w2[(size_t)k * DD + n], acc);
  state[idx] = acc;
}

// ---------------------------------------------------------------------------
// K6: trace shift + trace MLP; writes both outputs in return order.
//   out[0 : B*D)                       = next_activated_zL
//   out[B*D : B*D + B*D*M)             = next_zL_trace
// ---------------------------------------------------------------------------
__global__ __launch_bounds__(256) void trace_kernel(
    const float* __restrict__ zL_trace, const float* __restrict__ state,
    const float* __restrict__ tr_w1, const float* __restrict__ tr_b1,
    const float* __restrict__ tr_w2, const float* __restrict__ tr_b2,
    float* __restrict__ out) {
  __shared__ float w1[MM * TR_H];
  __shared__ float w2[TR_H];
  __shared__ float b1s[TR_H];
  int tid = threadIdx.x;
  for (int i = tid; i < MM * TR_H; i += 256) w1[i] = tr_w1[i];
  if (tid < TR_H) { w2[tid] = tr_w2[tid]; b1s[tid] = tr_b1[tid]; }
  __syncthreads();

  int idx = blockIdx.x * 256 + tid;                   // B*D
  float t[MM];
  const float* tr = zL_trace + (size_t)idx * MM;
  for (int m = 0; m < MM - 1; ++m) t[m] = tr[m + 1];
  t[MM - 1] = state[idx];

  float* otr = out + (size_t)BB * DD + (size_t)idx * MM;
  for (int m = 0; m < MM; ++m) otr[m] = t[m];

  float o = tr_b2[0];
  for (int hh = 0; hh < TR_H; ++hh) {
    float a = b1s[hh];
    for (int m = 0; m < MM; ++m) a = fmaf(t[m], w1[m * TR_H + hh], a);
    o = fmaf(fmaxf(a, 0.f), w2[hh], o);
  }
  out[idx] = o;
}

// ---------------------------------------------------------------------------
// Launch
// ---------------------------------------------------------------------------
extern "C" void kernel_launch(void* const* d_in, const int* in_sizes, int n_in,
                              void* d_out, int out_size, void* d_ws, size_t ws_size,
                              hipStream_t stream) {
  const float* activated_zL = (const float*)d_in[0];
  const float* zL_trace     = (const float*)d_in[1];
  const float* zH           = (const float*)d_in[2];
  const float* x_context    = (const float*)d_in[3];
  const float* sync_action  = (const float*)d_in[4];
  const float* qp_w  = (const float*)d_in[5];
  const float* qp_b  = (const float*)d_in[6];
  const float* wq    = (const float*)d_in[7];
  const float* bq    = (const float*)d_in[8];
  const float* wk    = (const float*)d_in[9];
  const float* bk    = (const float*)d_in[10];
  const float* wv    = (const float*)d_in[11];
  const float* bv    = (const float*)d_in[12];
  const float* wo    = (const float*)d_in[13];
  const float* bo    = (const float*)d_in[14];
  const float* syn_w1 = (const float*)d_in[15];
  const float* syn_b1 = (const float*)d_in[16];
  const float* syn_w2 = (const float*)d_in[17];
  const float* syn_b2 = (const float*)d_in[18];
  const float* tr_w1 = (const float*)d_in[19];
  const float* tr_b1 = (const float*)d_in[20];
  const float* tr_w2 = (const float*)d_in[21];
  const float* tr_b2 = (const float*)d_in[22];

  char* ws = (char*)d_ws;
  float*  qvec     = (float*)(ws + OFF_QVEC);
  float*  pq       = (float*)(ws + OFF_PQ);
  bf16_t* wkb      = (bf16_t*)(ws + OFF_WKB);
  bf16_t* wvb      = (bf16_t*)(ws + OFF_WVB);
  float*  kv_state = (float*)(ws + OFF_KVS);
  float*  ksum     = (float*)(ws + OFF_KSUM);
  float*  attn     = (float*)(ws + OFF_ATTN);
  float*  attn_out = (float*)(ws + OFF_AO);
  float*  h1       = (float*)(ws + OFF_H1);
  float*  state    = (float*)(ws + OFF_ST);
  float*  out      = (float*)d_out;

  qproj_kernel<<<BB * EE / 256, 256, 0, stream>>>(sync_action, qp_w, qp_b, qvec);
  pq_kernel<<<BB * EE / 256, 256, 0, stream>>>(qvec, wq, bq, pq);
  conv_weight_kernel<<<1024, 128, 0, stream>>>(wk, wv, wkb, wvb);
  kv_state_kernel<<<BB * (HEADS / 2), 256, 0, stream>>>(x_context, zH, bk, bv,
                                                        wkb, wvb, kv_state, ksum);
  attn_kernel<<<BB * EE / 256, 256, 0, stream>>>(pq, kv_state, ksum, attn);
  attnout_kernel<<<BB * EE / 256, 256, 0, stream>>>(attn, wo, bo, attn_out);
  syn1_kernel<<<BB * SYN_H / 256, 256, 0, stream>>>(attn_out, activated_zL, syn_w1,
                                                    syn_b1, h1);
  syn2_kernel<<<BB * DD / 256, 256, 0, stream>>>(h1, syn_w2, syn_b2, state);
  trace_kernel<<<BB * DD / 256, 256, 0, stream>>>(zL_trace, state, tr_w1, tr_b1,
                                                  tr_w2, tr_b2, out);
}
